// _Mamba2Block_41790031790103
// MI455X (gfx1250) — matrix operand order repaired and checked by reading, not compiled, not run on hardware
//
#include <hip/hip_runtime.h>

// ---------------------------------------------------------------------------
// Mamba2 block forward for MI455X (gfx1250, wave32).
//   - bf16 WMMA (v_wmma_f32_16x16x32_bf16) for in_proj / out_proj GEMMs
//   - Tensor Data Mover stages all GEMM tiles (double-buffered: DMA of K-step
//     i+1 overlaps WMMA of K-step i, s_wait_tensorcnt 2 drains only the old
//     pair) and prefetches the scan's next-step B/C/x rows behind the VALU
//     work of the current step.
//   - register-resident 64x64 f32 state per (batch, head) for the scan
// ---------------------------------------------------------------------------

typedef __bf16 bf16_t;
typedef __attribute__((ext_vector_type(16))) __bf16 v16bf;
typedef __attribute__((ext_vector_type(8)))  __bf16 v8bf;
typedef __attribute__((ext_vector_type(8)))  float  v8f;
typedef __attribute__((ext_vector_type(4)))  unsigned int u32x4;
typedef __attribute__((ext_vector_type(8)))  int i32x8;
typedef __attribute__((ext_vector_type(4)))  int i32x4;

#define D_MODEL  256
#define D_STATE  64
#define D_CONV   4
#define HEADDIM  64
#define D_INNER  512
#define NHEADS   8
#define CONV_DIM 640            // D_INNER + 2*D_STATE
#define DPROJ    1160           // 2*D_INNER + 2*D_STATE + NHEADS
#define BATCH    8
#define SEQ      4096
#define MTOK     (BATCH*SEQ)    // 32768 tokens

// GEMM tiling: 128x64 block tile, 8 waves (4M x 2N), 2x2 WMMA tiles per wave.
#define MT    128
#define NT    64
#define KT    32
#define LDSTR 40                // bf16 LDS row stride (80B) == TDM pad layout
#define SA_BYTES (MT * LDSTR * 2)           // 10240
#define SB_BYTES (NT * LDSTR * 2)           // 5120
#define GEMM_BUF_BYTES (SA_BYTES + SB_BYTES)
#define GEMM_LDS_BYTES (2 * GEMM_BUF_BYTES) // double buffered

// Scan dynamic-LDS layout (bytes): BC[2][128]f32 | X[2][64]f32 | dt[2]
#define SCAN_BC_OFF   0
#define SCAN_X_OFF    1024
#define SCAN_DT_OFF   1536
#define SCAN_LDS_BYTES 2048

#if __has_builtin(__builtin_amdgcn_tensor_load_to_lds) && \
    __has_builtin(__builtin_amdgcn_s_wait_tensorcnt)
#define HAS_TDM 1
#else
#define HAS_TDM 0
#endif

__device__ __forceinline__ float silu_f(float v) {
    return v / (1.f + __expf(-v));
}

#if HAS_TDM
// ---------------------------------------------------------------------------
// Issue one TDM 2-D tile load: tile (tile0 x tile1) elements from a row-major
// tensor with row stride stride0 elements, into LDS at byte offset lds_off.
// dsz_log2: 0=1B,1=2B,2=4B,3=8B. flags_d0: pre-shifted pad_* bits for g1[0].
// Descriptor layout per CDNA5 ISA section 8.3 / 8.4.
// ---------------------------------------------------------------------------
__device__ __forceinline__ void tdm_load_2d(
    unsigned lds_off, const void* gptr,
    unsigned tdim0, unsigned tdim1,
    unsigned tile0, unsigned tile1,
    unsigned long long stride0_elems,
    unsigned dsz_log2, unsigned flags_d0) {
    unsigned long long ga = (unsigned long long)(uintptr_t)gptr;
    u32x4 g0;
    g0[0] = 1u;                                   // count=1, user desc
    g0[1] = lds_off;                              // lds_addr (bytes)
    g0[2] = (unsigned)ga;                         // global_addr[31:0]
    g0[3] = (unsigned)((ga >> 32) & 0x1FFFFFFu)   // global_addr[56:32]
          | 0x80000000u;                          // type=2 (bits 127:126)
    i32x8 g1;
    g1[0] = (int)((dsz_log2 << 16) | flags_d0);
    g1[1] = (int)((tdim0 & 0xFFFFu) << 16);                     // dim0[15:0]
    g1[2] = (int)((tdim0 >> 16) | ((tdim1 & 0xFFFFu) << 16));   // dim0 hi|dim1 lo
    g1[3] = (int)((tdim1 >> 16) | (tile0 << 16));               // dim1 hi|tile0
    g1[4] = (int)(tile1 & 0xFFFFu);                             // tile1 (tile2=0)
    g1[5] = (int)(unsigned)stride0_elems;                       // stride0[31:0]
    g1[6] = (int)(unsigned)(stride0_elems >> 32);               // stride0[47:32]
    g1[7] = 0;
    i32x4 z4 = {0, 0, 0, 0};
#if __clang_major__ >= 23
    i32x8 z8 = {0, 0, 0, 0, 0, 0, 0, 0};
    __builtin_amdgcn_tensor_load_to_lds(g0, g1, z4, z4, z8, 0);
#else
    __builtin_amdgcn_tensor_load_to_lds(g0, g1, z4, z4, 0);
#endif
}
// pad every 16 DWORDs (one 32-bf16 row) with 4 DWORDs -> 40-bf16 row stride
#define TDM_PAD_BF16_ROW ((1u << 20) | (3u << 22) | (3u << 25))
#endif  // HAS_TDM

// ---------------------------------------------------------------------------
// fp32 -> bf16 weight conversion (one-time per call, tiny)
// ---------------------------------------------------------------------------
__global__ __launch_bounds__(256) void f32_to_bf16_kernel(
    const float* __restrict__ in, bf16_t* __restrict__ out, int n) {
    int i = blockIdx.x * 256 + threadIdx.x;
    if (i < n) out[i] = (bf16_t)in[i];
}

// ---------------------------------------------------------------------------
// Input RMSNorm over D_MODEL=256, emit bf16 activations.
// One wave per row: 8 floats/lane, shuffle reduction (wave32).
// ---------------------------------------------------------------------------
__global__ __launch_bounds__(256) void rmsnorm_in_kernel(
    const float* __restrict__ x, const float* __restrict__ w,
    bf16_t* __restrict__ out) {
    int row = blockIdx.x * 8 + (threadIdx.x >> 5);
    int l   = threadIdx.x & 31;
    const float* xr = x + (size_t)row * D_MODEL;
    float4 a = ((const float4*)xr)[l * 2];
    float4 b = ((const float4*)xr)[l * 2 + 1];
    float ss = a.x*a.x + a.y*a.y + a.z*a.z + a.w*a.w
             + b.x*b.x + b.y*b.y + b.z*b.z + b.w*b.w;
    for (int off = 16; off; off >>= 1) ss += __shfl_xor(ss, off, 32);
    float s = rsqrtf(ss * (1.f / 256.f) + 1.1920929e-7f);
    int c = l * 8;
    bf16_t* orow = out + (size_t)row * D_MODEL;
    orow[c + 0] = (bf16_t)(a.x * s * w[c + 0]);
    orow[c + 1] = (bf16_t)(a.y * s * w[c + 1]);
    orow[c + 2] = (bf16_t)(a.z * s * w[c + 2]);
    orow[c + 3] = (bf16_t)(a.w * s * w[c + 3]);
    orow[c + 4] = (bf16_t)(b.x * s * w[c + 4]);
    orow[c + 5] = (bf16_t)(b.y * s * w[c + 5]);
    orow[c + 6] = (bf16_t)(b.z * s * w[c + 6]);
    orow[c + 7] = (bf16_t)(b.w * s * w[c + 7]);
}

// ---------------------------------------------------------------------------
// TN GEMM with bf16 WMMA:  C[M,N] = A[M,K] * B[N,K]^T  (+ optional residual)
// Double-buffered TDM pipeline: wave 0 issues the next K-step's tile pair into
// the alternate LDS buffer, then waits TENSORcnt<=2 (old pair only). TDM OOB
// zero-fills the N tail; TDM padding produces the 40-bf16 row stride.
// ---------------------------------------------------------------------------
__global__ __launch_bounds__(256) void wmma_gemm_tn_kernel(
    const bf16_t* __restrict__ A, const bf16_t* __restrict__ B,
    float* __restrict__ C, const float* __restrict__ resid,
    int M, int N, int K) {
    extern __shared__ char smem_raw[];

    int tid = threadIdx.x;
    int m0 = blockIdx.y * MT;
    int n0 = blockIdx.x * NT;
    int w  = tid >> 5, l = tid & 31;
    int wm = (w & 3) * 32;      // wave M offset within block tile
    int wn = (w >> 2) * 32;     // wave N offset within block tile
    int lr = l & 15;            // lane row/col within 16
    int lh = l >> 4;            // lane half selector

    v8f zero = {0.f,0.f,0.f,0.f,0.f,0.f,0.f,0.f};
    v8f acc[2][2];
    acc[0][0] = zero; acc[0][1] = zero; acc[1][0] = zero; acc[1][1] = zero;

#if HAS_TDM
    // prologue: DMA first tile pair into buffer 0
    if (w == 0) {
        tdm_load_2d(0u, A + (size_t)m0 * K, (unsigned)K, (unsigned)(M - m0),
                    KT, MT, (unsigned long long)K, 1u, TDM_PAD_BF16_ROW);
        tdm_load_2d(SA_BYTES, B + (size_t)n0 * K, (unsigned)K, (unsigned)(N - n0),
                    KT, NT, (unsigned long long)K, 1u, TDM_PAD_BF16_ROW);
    }
    int buf = 0;
#else
    int arow = tid >> 1, aseg = (tid & 1) * 16;
    int brow = tid >> 2, bseg = (tid & 3) * 8;
#endif

    for (int kk = 0; kk < K; kk += KT) {
#if HAS_TDM
        bf16_t* sA = (bf16_t*)(smem_raw + buf * GEMM_BUF_BYTES);
        bf16_t* sB = (bf16_t*)(smem_raw + buf * GEMM_BUF_BYTES + SA_BYTES);
        if (w == 0) {
            int kn = kk + KT;
            if (kn < K) {
                unsigned nb = (unsigned)((buf ^ 1) * GEMM_BUF_BYTES);
                tdm_load_2d(nb, A + (size_t)m0 * K + kn,
                            (unsigned)(K - kn), (unsigned)(M - m0),
                            KT, MT, (unsigned long long)K, 1u, TDM_PAD_BF16_ROW);
                tdm_load_2d(nb + SA_BYTES, B + (size_t)n0 * K + kn,
                            (unsigned)(K - kn), (unsigned)(N - n0),
                            KT, NT, (unsigned long long)K, 1u, TDM_PAD_BF16_ROW);
                __builtin_amdgcn_s_wait_tensorcnt(2);  // current pair done
            } else {
                __builtin_amdgcn_s_wait_tensorcnt(0);
            }
        }
#else
        bf16_t* sA = (bf16_t*)smem_raw;
        bf16_t* sB = (bf16_t*)(smem_raw + SA_BYTES);
        const bf16_t* ag = A + (size_t)(m0 + arow) * K + kk + aseg;
        *(v8bf*)&sA[arow * LDSTR + aseg]     = *(const v8bf*)ag;
        *(v8bf*)&sA[arow * LDSTR + aseg + 8] = *(const v8bf*)(ag + 8);
        v8bf bv = {};
        if (n0 + brow < N)
            bv = *(const v8bf*)(B + (size_t)(n0 + brow) * K + kk + bseg);
        *(v8bf*)&sB[brow * LDSTR + bseg] = bv;
        if (kk + KT < K) __builtin_prefetch(ag + KT, 0, 0);
#endif
        __syncthreads();   // publish current buffer to all waves

#pragma unroll
        for (int mi = 0; mi < 2; ++mi) {
            // A fragment, 16x32 bf16 (ISA 7.12.2): lane half 0 -> K 0-7 & 16-23,
            // lane half 1 -> K 8-15 & 24-31, row = M
            int row = wm + mi * 16 + lr;
            union { v16bf v; v8bf h[2]; } af;
            af.h[0] = *(const v8bf*)&sA[row * LDSTR + lh * 8];
            af.h[1] = *(const v8bf*)&sA[row * LDSTR + lh * 8 + 16];
#pragma unroll
            for (int ni = 0; ni < 2; ++ni) {
                // B fragment, 32x16 bf16: lane = column (W row), lane half
                // 0 -> K 0-7 & 16-23, lane half 1 -> K 8-15 & 24-31.
                int col = wn + ni * 16 + lr;
                union { v16bf v; v8bf h[2]; } bfrag;
                bfrag.h[0] = *(const v8bf*)&sB[col * LDSTR + lh * 8];
                bfrag.h[1] = *(const v8bf*)&sB[col * LDSTR + lh * 8 + 16];
                acc[mi][ni] = __builtin_amdgcn_wmma_f32_16x16x32_bf16(
                    false, af.v, false, bfrag.v, (short)0, acc[mi][ni],
                    false, false);
            }
        }
        __syncthreads();   // all reads done; in-flight DMA may own other buffer
#if HAS_TDM
        buf ^= 1;
#endif
    }

    // C/D layout: VGPR v, lane half lh -> row M = v + 8*lh, col N = lane%16
#pragma unroll
    for (int mi = 0; mi < 2; ++mi) {
#pragma unroll
        for (int ni = 0; ni < 2; ++ni) {
            int cb = n0 + wn + ni * 16 + lr;
            if (cb >= N) continue;
            int rb = m0 + wm + mi * 16 + lh * 8;
#pragma unroll
            for (int v = 0; v < 8; ++v) {
                size_t idx = (size_t)(rb + v) * N + cb;
                float val = acc[mi][ni][v];
                if (resid) val += resid[idx];
                C[idx] = val;
            }
        }
    }
}

// ---------------------------------------------------------------------------
// Depthwise causal conv (k=4) over xBC channels + SiLU. Pure bandwidth.
// ---------------------------------------------------------------------------
__global__ __launch_bounds__(256) void conv_silu_kernel(
    const float* __restrict__ zxbcdt, const float* __restrict__ cw,
    const float* __restrict__ cb, float* __restrict__ xBCc) {
    int idx = blockIdx.x * 256 + threadIdx.x;
    int c = idx % CONV_DIM;
    int m = idx / CONV_DIM;
    int t = m & (SEQ - 1);
    float acc = cb[c];
#pragma unroll
    for (int k = 0; k < D_CONV; ++k) {
        int tt = t + k - (D_CONV - 1);
        if (tt >= 0)
            acc += zxbcdt[(size_t)(m + k - (D_CONV - 1)) * DPROJ + D_INNER + c]
                 * cw[c * D_CONV + k];
    }
    xBCc[(size_t)m * CONV_DIM + c] = silu_f(acc);
}

// ---------------------------------------------------------------------------
// Selective scan: one workgroup per (batch, head). 64x64 f32 state lives in
// registers (16/thread). TDM prefetches step t+1's B|C (512B) and x (256B)
// rows into the alternate LDS buffer while all waves compute step t; the
// TENSORcnt wait lands after the step's VALU work. dt(t+1) is computed in a
// register by lane 32 and published in the same rotation.
// ---------------------------------------------------------------------------
__global__ __launch_bounds__(256) void scan_kernel(
    const float* __restrict__ zxbcdt, const float* __restrict__ xBCc,
    const float* __restrict__ dt_bias, const float* __restrict__ A_log,
    const float* __restrict__ Dp, float* __restrict__ yraw) {
    extern __shared__ char sm[];
    float* sBC = (float*)(sm + SCAN_BC_OFF);   // [2][128]: B | C
    float* sX  = (float*)(sm + SCAN_X_OFF);    // [2][64]
    float* sdt = (float*)(sm + SCAN_DT_OFF);   // [2]

    int wg = blockIdx.x;
    int h = wg & (NHEADS - 1);
    int b = wg >> 3;
    int tid = threadIdx.x;
    int w = tid >> 5;
    int p = tid >> 2;           // head-dim row 0..63
    int q = tid & 3;            // quad slot -> state cols [q*16, q*16+16)
    int nb = q * 16;

    float st[16];
#pragma unroll
    for (int j = 0; j < 16; ++j) st[j] = 0.f;

    float dtb   = dt_bias[h];
    float negA  = -__expf(A_log[h]);
    float dcoef = Dp[h];

    const float* xbase  = xBCc + (size_t)b * SEQ * CONV_DIM;
    const float* dtbase = zxbcdt + (size_t)b * SEQ * DPROJ + (DPROJ - NHEADS) + h;

#if HAS_TDM
    if (w == 0) {
        tdm_load_2d(SCAN_BC_OFF, xbase + D_INNER, 128, 1, 128, 1, 128ull, 2u, 0u);
        tdm_load_2d(SCAN_X_OFF,  xbase + h * HEADDIM, 64, 1, 64, 1, 64ull, 2u, 0u);
        __builtin_amdgcn_s_wait_tensorcnt(0);
    }
    if (tid == 32) {
        float dr = dtbase[0] + dtb;
        sdt[0] = (dr > 20.f) ? dr : __logf(1.f + __expf(dr));
    }
    __syncthreads();
#endif

    for (int t = 0; t < SEQ; ++t) {
        int cur = t & 1, nxt = cur ^ 1;
#if HAS_TDM
        bool hn = (t + 1) < SEQ;
        if (w == 0 && hn) {
            const float* nrow = xbase + (size_t)(t + 1) * CONV_DIM;
            tdm_load_2d(SCAN_BC_OFF + nxt * 512, nrow + D_INNER,
                        128, 1, 128, 1, 128ull, 2u, 0u);
            tdm_load_2d(SCAN_X_OFF + nxt * 256, nrow + h * HEADDIM,
                        64, 1, 64, 1, 64ull, 2u, 0u);
        }
        float dtn = 0.f;
        if (tid == 32 && hn) {
            float dr = dtbase[(size_t)(t + 1) * DPROJ] + dtb;
            dtn = (dr > 20.f) ? dr : __logf(1.f + __expf(dr));
        }
#else
        const float* xrow = xbase + (size_t)t * CONV_DIM;
        if (tid < 128)       sBC[cur * 128 + tid] = xrow[D_INNER + tid];
        else if (tid < 192)  sX[cur * 64 + (tid - 128)] = xrow[h * HEADDIM + (tid - 128)];
        else if (tid == 192) {
            float dr = dtbase[(size_t)t * DPROJ] + dtb;
            sdt[cur] = (dr > 20.f) ? dr : __logf(1.f + __expf(dr));
        }
        __syncthreads();
#endif
        const float* bc = sBC + cur * 128;      // B at [0,64), C at [64,128)
        float dt   = sdt[cur];
        float dA   = __expf(negA * dt);
        float xp   = sX[cur * 64 + p];
        float coef = dt * xp;
        float accv = 0.f;
#pragma unroll
        for (int j = 0; j < 16; ++j) {
            st[j] = st[j] * dA + coef * bc[nb + j];
            accv += st[j] * bc[64 + nb + j];
        }
        accv += __shfl_xor(accv, 1, 32);
        accv += __shfl_xor(accv, 2, 32);
        size_t m = (size_t)b * SEQ + t;
        if (q == 0)
            yraw[m * D_INNER + h * HEADDIM + p] = accv + dcoef * xp;
#if HAS_TDM
        if (w == 0 && hn) __builtin_amdgcn_s_wait_tensorcnt(0);
        if (tid == 32 && hn) sdt[nxt] = dtn;
#endif
        __syncthreads();
    }
}

// ---------------------------------------------------------------------------
// y * silu(z) then gated RMSNorm over D_INNER=512, emit bf16 for out_proj.
// One wave per row, 16 elems/lane.
// ---------------------------------------------------------------------------
__global__ __launch_bounds__(256) void gate_gnorm_kernel(
    const float* __restrict__ yraw, const float* __restrict__ zxbcdt,
    const float* __restrict__ gw, bf16_t* __restrict__ ybf) {
    int row = blockIdx.x * 8 + (threadIdx.x >> 5);
    int l   = threadIdx.x & 31;
    const float* yr = yraw + (size_t)row * D_INNER;
    const float* zr = zxbcdt + (size_t)row * DPROJ;   // z = first D_INNER cols
    float g[16];
    float ss = 0.f;
#pragma unroll
    for (int i = 0; i < 16; ++i) {
        int c = l * 16 + i;
        float v = yr[c] * silu_f(zr[c]);
        g[i] = v;
        ss += v * v;
    }
    for (int off = 16; off; off >>= 1) ss += __shfl_xor(ss, off, 32);
    float s = rsqrtf(ss * (1.f / 512.f) + 1e-5f);
    bf16_t* orow = ybf + (size_t)row * D_INNER;
#pragma unroll
    for (int i = 0; i < 16; ++i) {
        int c = l * 16 + i;
        orow[c] = (bf16_t)(g[i] * s * gw[c]);
    }
}

// ---------------------------------------------------------------------------
// launch
// ---------------------------------------------------------------------------
extern "C" void kernel_launch(void* const* d_in, const int* in_sizes, int n_in,
                              void* d_out, int out_size, void* d_ws, size_t ws_size,
                              hipStream_t stream) {
    (void)in_sizes; (void)n_in; (void)out_size; (void)ws_size;
    const float* x       = (const float*)d_in[0];
    const float* W_in    = (const float*)d_in[1];
    const float* conv_w  = (const float*)d_in[2];
    const float* conv_b  = (const float*)d_in[3];
    const float* dt_bias = (const float*)d_in[4];
    const float* A_log   = (const float*)d_in[5];
    const float* Dp      = (const float*)d_in[6];
    const float* gnorm_w = (const float*)d_in[7];
    const float* norm_w  = (const float*)d_in[8];
    const float* W_out   = (const float*)d_in[9];
    float* out = (float*)d_out;

    // workspace carve (all chunks 256B-aligned by construction)
    char* p = (char*)d_ws;
    bf16_t* xn_bf   = (bf16_t*)p; p += (size_t)MTOK * D_MODEL * sizeof(bf16_t);
    bf16_t* Win_bf  = (bf16_t*)p; p += (size_t)DPROJ * D_MODEL * sizeof(bf16_t);
    bf16_t* Wout_bf = (bf16_t*)p; p += (size_t)D_MODEL * D_INNER * sizeof(bf16_t);
    bf16_t* ybf     = (bf16_t*)p; p += (size_t)MTOK * D_INNER * sizeof(bf16_t);
    float*  zxbcdt  = (float*)p;  p += (size_t)MTOK * DPROJ * sizeof(float);
    float*  xBCc    = (float*)p;  p += (size_t)MTOK * CONV_DIM * sizeof(float);
    float*  yraw    = (float*)p;  p += (size_t)MTOK * D_INNER * sizeof(float);

    f32_to_bf16_kernel<<<(DPROJ * D_MODEL + 255) / 256, 256, 0, stream>>>(
        W_in, Win_bf, DPROJ * D_MODEL);
    f32_to_bf16_kernel<<<(D_MODEL * D_INNER + 255) / 256, 256, 0, stream>>>(
        W_out, Wout_bf, D_MODEL * D_INNER);

    rmsnorm_in_kernel<<<MTOK / 8, 256, 0, stream>>>(x, norm_w, xn_bf);

    // in_proj: [32768,1160] = xn[32768,256] * W_in[1160,256]^T
    wmma_gemm_tn_kernel<<<dim3((DPROJ + NT - 1) / NT, MTOK / MT), 256,
                          GEMM_LDS_BYTES, stream>>>(
        xn_bf, Win_bf, zxbcdt, nullptr, MTOK, DPROJ, D_MODEL);

    conv_silu_kernel<<<(MTOK * CONV_DIM) / 256, 256, 0, stream>>>(
        zxbcdt, conv_w, conv_b, xBCc);

    scan_kernel<<<BATCH * NHEADS, 256, SCAN_LDS_BYTES, stream>>>(
        zxbcdt, xBCc, dt_bias, A_log, Dp, yraw);

    gate_gnorm_kernel<<<MTOK / 8, 256, 0, stream>>>(yraw, zxbcdt, gnorm_w, ybf);

    // out_proj + residual: out[32768,256] = ybf * W_out[256,512]^T + x
    wmma_gemm_tn_kernel<<<dim3(D_MODEL / NT, MTOK / MT), 256,
                          GEMM_LDS_BYTES, stream>>>(
        ybf, Wout_bf, out, x, MTOK, D_MODEL, D_INNER);
}
